// VectorQuantization_54485955117336
// MI455X (gfx1250) — matrix operand order repaired and checked by reading, not compiled, not run on hardware
//
#include <hip/hip_runtime.h>

// ---------------------------------------------------------------------------
// VQ-VAE vector quantization for MI455X (gfx1250, wave32, WMMA).
//   N = 32768 rows, d = 64, K = 4096 codes.
//   argmin_k(||w_k||^2 - 2 x.w_k) via bf16 hi/lo-split WMMA (near-fp32),
//   codebook double-buffered through LDS with GLOBAL_LOAD_ASYNC_TO_LDS_B128,
//   then gather + fused loss = 1.25*(q-x)^2 with non-temporal output stores.
// ---------------------------------------------------------------------------

typedef __attribute__((ext_vector_type(16))) __bf16 v16bf;
typedef __attribute__((ext_vector_type(8)))  float  v8f;
typedef __attribute__((ext_vector_type(4)))  float  v4f;

#define NUM_EMB   4096
#define DIM       64
#define ROWS      32768                 // 8 * 4096
#define CHUNK     128                   // codes per LDS buffer
#define NCHUNK    (NUM_EMB / CHUNK)     // 32
#define WAVES_PB  8
#define ROWS_PB   (WAVES_PB * 16)       // 128 rows per workgroup

// ---- CDNA5 async global->LDS copy (ASYNCcnt-tracked), inline asm form ------
__device__ __forceinline__ void async_b128(unsigned lds_addr, const void* gaddr) {
  asm volatile("global_load_async_to_lds_b128 %0, %1, off"
               :: "v"(lds_addr), "v"(gaddr)
               : "memory");
}
__device__ __forceinline__ void wait_async_le8() {
  asm volatile("s_wait_asynccnt 0x8" ::: "memory");
}
__device__ __forceinline__ void wait_async_0() {
  asm volatile("s_wait_asynccnt 0x0" ::: "memory");
}

// ---------------------------------------------------------------------------
// Kernel 1: codebook prep — bf16 hi/lo split planes + per-code squared norm.
// ---------------------------------------------------------------------------
__global__ __launch_bounds__(256) void vq_prep(const float* __restrict__ w,
                                               float* __restrict__ wsq,
                                               unsigned short* __restrict__ whi_u,
                                               unsigned short* __restrict__ wlo_u) {
  __bf16* whi = (__bf16*)whi_u;
  __bf16* wlo = (__bf16*)wlo_u;
  int c = blockIdx.x * 256 + threadIdx.x;
  if (c >= NUM_EMB) return;
  float s = 0.0f;
  #pragma unroll 8
  for (int d = 0; d < DIM; ++d) {
    float v = w[c * DIM + d];
    s += v * v;
    __bf16 hi = (__bf16)v;                    // RNE to bf16
    __bf16 lo = (__bf16)(v - (float)hi);      // residual plane
    whi[c * DIM + d] = hi;
    wlo[c * DIM + d] = lo;
  }
  wsq[c] = s;
}

// ---------------------------------------------------------------------------
// Kernel 2: WMMA argmin with async double-buffered codebook staging.
// ---------------------------------------------------------------------------
__global__ __launch_bounds__(256) void vq_argmin(const float* __restrict__ x,
                                                 const float* __restrict__ wsq,
                                                 const unsigned short* __restrict__ whi_u,
                                                 const unsigned short* __restrict__ wlo_u,
                                                 int* __restrict__ idx_out) {
  // [buffer][hi|lo][CHUNK*DIM] : 4 x 16 KB = 64 KB
  __shared__ __align__(16) __bf16 sh[2][2][CHUNK * DIM];

  const __bf16* whi = (const __bf16*)whi_u;
  const __bf16* wlo = (const __bf16*)wlo_u;

  const int tid  = threadIdx.x;
  const int wave = tid >> 5;
  const int lane = tid & 31;
  const int r    = lane & 15;   // A-matrix row M (and B-matrix column N)
  const int h    = lane >> 4;   // half-wave selector (K-slice picker)
  const int waveRowBase = blockIdx.x * ROWS_PB + wave * 16;

  // ---- A fragments: x rows split into bf16 hi/lo, 16x32 A, contiguous k = 16h + s ----
  v16bf a_hi[2], a_lo[2];
  const float* xrow = x + (size_t)(waveRowBase + r) * DIM;
  #pragma unroll
  for (int f = 0; f < 2; ++f) {
    #pragma unroll
    for (int t = 0; t < 2; ++t) {
      const v4f* p = (const v4f*)(xrow + f * 32 + h * 16 + t * 8);
      v4f u0 = p[0], u1 = p[1];
      float vals[8] = {u0.x, u0.y, u0.z, u0.w, u1.x, u1.y, u1.z, u1.w};
      #pragma unroll
      for (int j = 0; j < 8; ++j) {
        __bf16 hi = (__bf16)vals[j];
        __bf16 lo = (__bf16)(vals[j] - (float)hi);
        a_hi[f][t * 8 + j] = hi;
        a_lo[f][t * 8 + j] = lo;
      }
    }
  }

  float bestd[8];
  int   besti[8];
  #pragma unroll
  for (int v = 0; v < 8; ++v) { bestd[v] = 3.0e38f; besti[v] = 0; }

  // ---- async stage of one chunk (hi+lo planes): 8 b128 issues per thread ---
  auto stage = [&](int buf, int chunk) {
    const char* ghi = (const char*)(whi + (size_t)chunk * CHUNK * DIM);
    const char* glo = (const char*)(wlo + (size_t)chunk * CHUNK * DIM);
    unsigned lhi = (unsigned)(size_t)&sh[buf][0][0];
    unsigned llo = (unsigned)(size_t)&sh[buf][1][0];
    #pragma unroll
    for (int k = 0; k < 4; ++k) {
      int off = (tid + k * 256) * 16;          // covers 16 KB per plane
      async_b128(lhi + off, ghi + off);
      async_b128(llo + off, glo + off);
    }
  };

  stage(0, 0);                                  // prologue fill of buffer 0

  for (int chunk = 0; chunk < NCHUNK; ++chunk) {
    const int buf = chunk & 1;
    if (chunk + 1 < NCHUNK) {
      stage(buf ^ 1, chunk + 1);                // fill other buffer in flight
      wait_async_le8();                         // current chunk's 8 are done
    } else {
      wait_async_0();
    }
    __syncthreads();                            // publish LDS to all waves

    const __bf16* base_hi = &sh[buf][0][0];
    const __bf16* base_lo = &sh[buf][1][0];

    #pragma unroll 2
    for (int t16 = 0; t16 < CHUNK / 16; ++t16) {
      // ---- B fragments: lane holds column (code) r, K-half by h ----
      const int c = t16 * 16 + r;
      const __bf16* bb_hi = base_hi + c * DIM + h * 16;
      const __bf16* bb_lo = base_lo + c * DIM + h * 16;
      v16bf b_hi0 = *(const v16bf*)(bb_hi);        // K 0..31 slice
      v16bf b_hi1 = *(const v16bf*)(bb_hi + 32);   // K 32..63 slice
      v16bf b_lo0 = *(const v16bf*)(bb_lo);
      v16bf b_lo1 = *(const v16bf*)(bb_lo + 32);

      v8f acc = {};
      acc = __builtin_amdgcn_wmma_f32_16x16x32_bf16(false, a_hi[0], false, b_hi0, (short)0, acc, false, false);
      acc = __builtin_amdgcn_wmma_f32_16x16x32_bf16(false, a_hi[1], false, b_hi1, (short)0, acc, false, false);
      acc = __builtin_amdgcn_wmma_f32_16x16x32_bf16(false, a_lo[0], false, b_hi0, (short)0, acc, false, false);
      acc = __builtin_amdgcn_wmma_f32_16x16x32_bf16(false, a_lo[1], false, b_hi1, (short)0, acc, false, false);
      acc = __builtin_amdgcn_wmma_f32_16x16x32_bf16(false, a_hi[0], false, b_lo0, (short)0, acc, false, false);
      acc = __builtin_amdgcn_wmma_f32_16x16x32_bf16(false, a_hi[1], false, b_lo1, (short)0, acc, false, false);

      // C/D layout: acc[v] = dot(row v+8h, code c). dist = |w|^2 - 2*dot.
      const int code = chunk * CHUNK + t16 * 16 + r;
      const float wq = wsq[code];
      #pragma unroll
      for (int v = 0; v < 8; ++v) {
        float dist = wq - 2.0f * acc[v];
        if (dist < bestd[v]) { bestd[v] = dist; besti[v] = code; }
      }
    }
    __syncthreads();   // all waves done reading buf before it is refilled
  }

  // ---- min-reduce across the 16 code-lanes (same rows, different codes) ----
  #pragma unroll
  for (int v = 0; v < 8; ++v) {
    float d = bestd[v];
    int   i = besti[v];
    #pragma unroll
    for (int m = 1; m <= 8; m <<= 1) {
      float od = __shfl_xor(d, m, 32);
      int   oi = __shfl_xor(i, m, 32);
      if (od < d || (od == d && oi < i)) { d = od; i = oi; }  // first-min ties
    }
    bestd[v] = d;
    besti[v] = i;
  }
  if (r == 0) {
    #pragma unroll
    for (int v = 0; v < 8; ++v)
      idx_out[waveRowBase + h * 8 + v] = besti[v];
  }
}

// ---------------------------------------------------------------------------
// Kernel 3: gather fp32 codebook row + fused loss = 1.25*(q-x)^2.
// ext-vector float4 I/O, non-temporal stores (outputs write-once, 16 MiB).
// ---------------------------------------------------------------------------
__global__ __launch_bounds__(256) void vq_out(const float* __restrict__ x,
                                              const float* __restrict__ w,
                                              const int* __restrict__ idx,
                                              float* __restrict__ outq,
                                              float* __restrict__ outl) {
  int t   = blockIdx.x * 256 + threadIdx.x;   // float4 index
  int row = t >> 4;                           // DIM/4 = 16 float4 per row
  int d4  = t & 15;
  int c   = idx[row];
  v4f xv = ((const v4f*)x)[t];
  v4f wv = ((const v4f*)w)[c * 16 + d4];
  v4f dv = wv - xv;
  v4f lv = 1.25f * dv * dv;
  __builtin_nontemporal_store(wv, (v4f*)outq + t);
  __builtin_nontemporal_store(lv, (v4f*)outl + t);
}

// ---------------------------------------------------------------------------
extern "C" void kernel_launch(void* const* d_in, const int* in_sizes, int n_in,
                              void* d_out, int out_size, void* d_ws, size_t ws_size,
                              hipStream_t stream) {
  const float* x = (const float*)d_in[0];   // [8,4096,64] f32
  const float* w = (const float*)d_in[1];   // [4096,64] f32
  float* out = (float*)d_out;               // quantized ++ loss

  // workspace layout (16B-aligned sections):
  //   wsq : 4096 f32 | idx : 32768 i32 | whi : 4096*64 bf16 | wlo : same
  char* ws = (char*)d_ws;
  float*          wsq = (float*)ws;
  int*            idx = (int*)(ws + (size_t)NUM_EMB * 4);
  unsigned short* whi = (unsigned short*)(ws + (size_t)NUM_EMB * 4 + (size_t)ROWS * 4);
  unsigned short* wlo = whi + (size_t)NUM_EMB * DIM;

  vq_prep<<<NUM_EMB / 256, 256, 0, stream>>>(w, wsq, whi, wlo);
  vq_argmin<<<ROWS / ROWS_PB, 256, 0, stream>>>(x, wsq, whi, wlo, idx);

  const int nq4 = ROWS * DIM / 4;           // 524288 float4 per output
  vq_out<<<nq4 / 256, 256, 0, stream>>>(x, w, idx, out, out + (size_t)ROWS * DIM);
}
